// MultiResourceAttention_72980084294339
// MI455X (gfx1250) — matrix operand order repaired and checked by reading, not compiled, not run on hardware
//
#include <hip/hip_runtime.h>

#define DM 1024
#define SEQ 2048
#define BATCH 4
#define ROWS (BATCH * SEQ)   // 8192

typedef __bf16 bf16_t;
typedef bf16_t v16bf __attribute__((ext_vector_type(16)));
typedef float  v8f   __attribute__((ext_vector_type(8)));

union AU { v16bf v; uint4 q[2]; };

__device__ __forceinline__ unsigned short f2bf(float f) {
    unsigned int u = __float_as_uint(f);
    u += 0x7FFFu + ((u >> 16) & 1u);     // round-to-nearest-even
    return (unsigned short)(u >> 16);
}

__device__ __forceinline__ v8f wmma_bf16(v16bf a, v16bf b, v8f c) {
    return __builtin_amdgcn_wmma_f32_16x16x32_bf16(
        /*neg_a=*/false, a, /*neg_b=*/false, b,
        /*c_mod=*/(short)0, c, /*reuse_a=*/false, /*reuse_b=*/false);
}

// CDNA5 async global->LDS copy (ASYNCcnt-tracked, no VGPR landing).
__device__ __forceinline__ void async_g2l_b128(unsigned lds_off, const void* gaddr) {
    asm volatile("global_load_async_to_lds_b128 %0, %1, off"
                 :: "v"(lds_off), "v"(gaddr) : "memory");
}
__device__ __forceinline__ void wait_asynccnt0() {
    asm volatile("s_wait_asynccnt 0x0" ::: "memory");
}

// ---------------------------------------------------------------------------
// Prep: x fp32 -> bf16 (row-major)
// ---------------------------------------------------------------------------
__global__ __launch_bounds__(256) void prep_x_kernel(const float* __restrict__ x,
                                                     unsigned short* __restrict__ xb) {
    const size_t n = (size_t)ROWS * DM;
    for (size_t i = (size_t)blockIdx.x * blockDim.x + threadIdx.x; i < n;
         i += (size_t)gridDim.x * blockDim.x) {
        xb[i] = f2bf(x[i]);
    }
}

// Prep: W[k][n] fp32 -> WT[n][k] bf16, for z = 0..2 (Wq, Wk, Wv)
__global__ __launch_bounds__(256) void prep_w_kernel(const float* __restrict__ Wq,
                                                     const float* __restrict__ Wk,
                                                     const float* __restrict__ Wv,
                                                     unsigned short* __restrict__ WT) {
    const int z = blockIdx.z;
    const float* W = (z == 0) ? Wq : ((z == 1) ? Wk : Wv);
    unsigned short* o = WT + (size_t)z * DM * DM;
    const size_t n = (size_t)DM * DM;
    for (size_t i = (size_t)blockIdx.x * blockDim.x + threadIdx.x; i < n;
         i += (size_t)gridDim.x * blockDim.x) {
        size_t k = i >> 10, c = i & (DM - 1);
        o[c * DM + k] = f2bf(W[i]);
    }
}

// ---------------------------------------------------------------------------
// QKV projection GEMM: out = x @ W + b  (bf16 in, f32 acc, bf16 out)
// grid = (ROWS/128, DM/128, 3); block = 256 (8 waves); wave = 16x128 tile
// z==2 stores V transposed: VT[b][d][s]
// ---------------------------------------------------------------------------
__global__ __launch_bounds__(256) void qkv_gemm_kernel(
        const unsigned short* __restrict__ xb,
        const unsigned short* __restrict__ WTs,
        const float* __restrict__ bq, const float* __restrict__ bk,
        const float* __restrict__ bv,
        unsigned short* __restrict__ Qb, unsigned short* __restrict__ Kb,
        unsigned short* __restrict__ VTb) {
    const int tid  = threadIdx.x;
    const int w    = tid >> 5;
    const int lane = tid & 31;
    const int laneh = lane & 15;
    const int hi   = (lane >= 16) ? 1 : 0;

    const int z  = blockIdx.z;
    const unsigned short* WT = WTs + (size_t)z * DM * DM;
    const float* bias = (z == 0) ? bq : ((z == 1) ? bk : bv);

    const int m0 = blockIdx.x * 128 + w * 16;
    const int n0 = blockIdx.y * 128;

    v8f c[8];
#pragma unroll
    for (int nf = 0; nf < 8; ++nf)
#pragma unroll
        for (int r = 0; r < 8; ++r) c[nf][r] = 0.0f;

    const unsigned short* Arow = xb + (size_t)(m0 + laneh) * DM;

    for (int k = 0; k < DM; k += 32) {
        AU a;
        a.q[0] = *(const uint4*)(Arow + k +      (hi ? 8 : 0));
        a.q[1] = *(const uint4*)(Arow + k + 16 + (hi ? 8 : 0));
#pragma unroll
        for (int nf = 0; nf < 8; ++nf) {
            const unsigned short* Wrow = WT + (size_t)(n0 + nf * 16 + laneh) * DM;
            AU bu; bu.q[0] = *(const uint4*)(Wrow + k + (hi ? 8 : 0)); bu.q[1] = *(const uint4*)(Wrow + k + 16 + (hi ? 8 : 0));
            v16bf bm = bu.v;
            c[nf] = wmma_bf16(a.v, bm, c[nf]);
        }
    }

#pragma unroll
    for (int nf = 0; nf < 8; ++nf) {
        const int n = n0 + nf * 16 + laneh;
        const float bi = bias[n];
#pragma unroll
        for (int r = 0; r < 8; ++r) {
            const int row = m0 + r + (hi ? 8 : 0);
            const unsigned short v = f2bf(c[nf][r] + bi);
            if (z == 0) {
                Qb[(size_t)row * DM + n] = v;
            } else if (z == 1) {
                Kb[(size_t)row * DM + n] = v;
            } else {
                const int b = row >> 11;          // /SEQ
                const int s = row & (SEQ - 1);
                VTb[((size_t)b * DM + n) * SEQ + s] = v;
            }
        }
    }
}

// ---------------------------------------------------------------------------
// Flash attention (16-row q tile, 64-key steps) + residual + LayerNorm.
// grid = BATCH * SEQ/16 = 512; block = 256 (8 waves). Wave w owns D-slice
// [w*128, (w+1)*128) for both score reduction and output columns.
// Residual x tile (64 KB fp32) is staged via async global->LDS copy that
// overlaps the whole K-loop.
// ---------------------------------------------------------------------------
__global__ __launch_bounds__(256) void flash_ln_kernel(
        const float* __restrict__ x,
        const float* __restrict__ gamma, const float* __restrict__ beta,
        const unsigned short* __restrict__ Qb,
        const unsigned short* __restrict__ Kb,
        const unsigned short* __restrict__ VTb,
        float* __restrict__ out) {
    __shared__ float x_lds[16 * DM];                       // 64 KB residual stage
    __shared__ float s_part[8][16][64];                    // 32 KB partial scores
    __shared__ float s_red[16][64];                        //  4 KB reduced scores
    __shared__ __align__(16) unsigned short p_lds[16][64]; //  2 KB P (bf16)
    __shared__ float m_lds[16], l_lds[16], sc_lds[16], sum_lds[16], ss_lds[16];

    const int tid  = threadIdx.x;
    const int w    = tid >> 5;
    const int lane = tid & 31;
    const int laneh = lane & 15;
    const int hi   = (lane >= 16) ? 1 : 0;
    const int hi8  = hi ? 8 : 0;

    const int b  = blockIdx.x >> 7;           // / (SEQ/16)
    const int q0 = (blockIdx.x & 127) * 16;

    // Kick off the async residual copy: 16 rows x 4 KB contiguous = 64 KB.
    {
        const char* xsrc = (const char*)(x + ((size_t)(b << 11) + q0) * DM);
        const unsigned lds_base = (unsigned)(unsigned long long)(&x_lds[0]);
#pragma unroll
        for (int it = 0; it < 16; ++it) {
            const int c16 = (tid + it * 256) * 16;        // byte offset 0..65520
            async_g2l_b128(lds_base + c16, xsrc + c16);
        }
    }

    if (tid < 16) { m_lds[tid] = -1e30f; l_lds[tid] = 0.0f; }

    // O accumulators: 8 frags of 16x16 over cols [w*128, w*128+128)
    v8f o[8];
#pragma unroll
    for (int nf = 0; nf < 8; ++nf)
#pragma unroll
        for (int r = 0; r < 8; ++r) o[nf][r] = 0.0f;

    // Preload Q A-fragments for this wave's 128-wide D slice (4 k-slices)
    const unsigned short* Qrow = Qb + (size_t)((b << 11) + q0 + laneh) * DM;
    AU qf[4];
#pragma unroll
    for (int ks = 0; ks < 4; ++ks) {
        const int dbase = w * 128 + ks * 32;
        qf[ks].q[0] = *(const uint4*)(Qrow + dbase +      (hi ? 8 : 0));
        qf[ks].q[1] = *(const uint4*)(Qrow + dbase + 16 + (hi ? 8 : 0));
    }
    __syncthreads();

    for (int kt = 0; kt < SEQ; kt += 64) {
        // ---- partial scores over this wave's D slice: S_part = Q * K^T ----
#pragma unroll
        for (int f = 0; f < 4; ++f) {
            v8f sc;
#pragma unroll
            for (int r = 0; r < 8; ++r) sc[r] = 0.0f;
            const int key = kt + f * 16 + laneh;
            const unsigned short* Krow = Kb + (size_t)((b << 11) + key) * DM;
#pragma unroll
            for (int ks = 0; ks < 4; ++ks) {
                AU bu; bu.q[0] = *(const uint4*)(Krow + w * 128 + ks * 32 + (hi ? 8 : 0)); bu.q[1] = *(const uint4*)(Krow + w * 128 + ks * 32 + 16 + (hi ? 8 : 0));
                v16bf bm = bu.v;
                sc = wmma_bf16(qf[ks].v, bm, sc);
            }
#pragma unroll
            for (int r = 0; r < 8; ++r)
                s_part[w][r + hi8][f * 16 + laneh] = sc[r];
        }
        __syncthreads();

        // ---- reduce partials across the 8 waves ----
        for (int e = tid; e < 1024; e += 256) {
            const int row = e >> 6, col = e & 63;
            float s = 0.0f;
#pragma unroll
            for (int ww = 0; ww < 8; ++ww) s += s_part[ww][row][col];
            s_red[row][col] = s;
        }
        __syncthreads();

        // ---- online softmax (one thread per query row) ----
        if (tid < 16) {
            const int row = tid;
            const float mold = m_lds[row];
            float mx = mold;
#pragma unroll
            for (int j = 0; j < 64; ++j) {
                const float sv = s_red[row][j] * 0.03125f;   // 1/sqrt(1024)
                mx = fmaxf(mx, sv);
            }
            float lsum = 0.0f;
#pragma unroll
            for (int j = 0; j < 64; ++j) {
                const float p = __expf(s_red[row][j] * 0.03125f - mx);
                lsum += p;
                p_lds[row][j] = f2bf(p);
            }
            const float scale = __expf(mold - mx);
            l_lds[row] = l_lds[row] * scale + lsum;
            m_lds[row] = mx;
            sc_lds[row] = scale;
        }
        __syncthreads();

        // ---- rescale O, then O += P * V (two 32-key chunks) ----
        float rs[8];
#pragma unroll
        for (int r = 0; r < 8; ++r) rs[r] = sc_lds[r + hi8];
#pragma unroll
        for (int nf = 0; nf < 8; ++nf)
#pragma unroll
            for (int r = 0; r < 8; ++r) o[nf][r] *= rs[r];

        AU pf[2];  // P as two 16x32 bf16 A-fragments (row stride 128 B)
#pragma unroll
        for (int c = 0; c < 2; ++c) {
            const char* prow = (const char*)&p_lds[laneh][0] + c * 64;
            pf[c].q[0] = *(const uint4*)(prow +      (hi ? 16 : 0));
            pf[c].q[1] = *(const uint4*)(prow + 32 + (hi ? 16 : 0));
        }

#pragma unroll
        for (int nf = 0; nf < 8; ++nf) {
            const int dcol = w * 128 + nf * 16 + laneh;
            const unsigned short* Vrow = VTb + ((size_t)b * DM + dcol) * SEQ;
#pragma unroll
            for (int c = 0; c < 2; ++c) {
                AU bu; bu.q[0] = *(const uint4*)(Vrow + kt + c * 32 + (hi ? 8 : 0)); bu.q[1] = *(const uint4*)(Vrow + kt + c * 32 + 16 + (hi ? 8 : 0));
                v16bf bm = bu.v;
                o[nf] = wmma_bf16(pf[c].v, bm, o[nf]);
            }
        }
        __syncthreads();
    }

    // ---- drain async residual copy before the epilogue consumes x_lds ----
    wait_asynccnt0();
    __syncthreads();

    // ---- epilogue: /l, residual from LDS, LayerNorm ----
    float il[8];
#pragma unroll
    for (int r = 0; r < 8; ++r) il[r] = 1.0f / l_lds[r + hi8];

#pragma unroll
    for (int nf = 0; nf < 8; ++nf) {
        const int col = w * 128 + nf * 16 + laneh;
#pragma unroll
        for (int r = 0; r < 8; ++r) {
            const int rowl = r + hi8;
            o[nf][r] = o[nf][r] * il[r] + x_lds[rowl * DM + col];   // h = x + attn
        }
    }

    if (tid < 16) { sum_lds[tid] = 0.0f; ss_lds[tid] = 0.0f; }
    __syncthreads();

    float ps[8], pq[8];
#pragma unroll
    for (int r = 0; r < 8; ++r) { ps[r] = 0.0f; pq[r] = 0.0f; }
#pragma unroll
    for (int nf = 0; nf < 8; ++nf)
#pragma unroll
        for (int r = 0; r < 8; ++r) { ps[r] += o[nf][r]; pq[r] += o[nf][r] * o[nf][r]; }
#pragma unroll
    for (int r = 0; r < 8; ++r) {
        atomicAdd(&sum_lds[r + hi8], ps[r]);
        atomicAdd(&ss_lds[r + hi8], pq[r]);
    }
    __syncthreads();

    float mu[8], rstd[8];
#pragma unroll
    for (int r = 0; r < 8; ++r) {
        mu[r] = sum_lds[r + hi8] * (1.0f / DM);
        const float var = ss_lds[r + hi8] * (1.0f / DM) - mu[r] * mu[r];
        rstd[r] = rsqrtf(var + 1e-5f);
    }
#pragma unroll
    for (int nf = 0; nf < 8; ++nf) {
        const int col = w * 128 + nf * 16 + laneh;
        const float g = gamma[col], be = beta[col];
#pragma unroll
        for (int r = 0; r < 8; ++r) {
            const int row = q0 + r + hi8;
            out[((size_t)(b << 11) + row) * DM + col] =
                (o[nf][r] - mu[r]) * rstd[r] * g + be;
        }
    }
}

// ---------------------------------------------------------------------------
extern "C" void kernel_launch(void* const* d_in, const int* in_sizes, int n_in,
                              void* d_out, int out_size, void* d_ws, size_t ws_size,
                              hipStream_t stream) {
    (void)in_sizes; (void)n_in; (void)out_size; (void)ws_size;
    const float* x     = (const float*)d_in[0];
    const float* Wq    = (const float*)d_in[1];
    const float* bq    = (const float*)d_in[2];
    const float* Wk    = (const float*)d_in[3];
    const float* bk    = (const float*)d_in[4];
    const float* Wv    = (const float*)d_in[5];
    const float* bv    = (const float*)d_in[6];
    const float* gamma = (const float*)d_in[7];
    const float* beta  = (const float*)d_in[8];
    float* out = (float*)d_out;

    char* ws = (char*)d_ws;
    // Workspace layout (bytes)
    const size_t SZ_XB = (size_t)ROWS * DM * 2;      // 16 MB
    const size_t SZ_WT = (size_t)3 * DM * DM * 2;    //  6 MB
    const size_t SZ_M  = (size_t)ROWS * DM * 2;      // 16 MB each
    unsigned short* xb  = (unsigned short*)(ws);
    unsigned short* WT  = (unsigned short*)(ws + SZ_XB);
    unsigned short* Qb  = (unsigned short*)(ws + SZ_XB + SZ_WT);
    unsigned short* Kb  = (unsigned short*)(ws + SZ_XB + SZ_WT + SZ_M);
    unsigned short* VTb = (unsigned short*)(ws + SZ_XB + SZ_WT + 2 * SZ_M);

    prep_x_kernel<<<4096, 256, 0, stream>>>(x, xb);
    prep_w_kernel<<<dim3(1024, 1, 3), 256, 0, stream>>>(Wq, Wk, Wv, WT);
    qkv_gemm_kernel<<<dim3(ROWS / 128, DM / 128, 3), 256, 0, stream>>>(
        xb, WT, bq, bk, bv, Qb, Kb, VTb);
    flash_ln_kernel<<<BATCH * (SEQ / 16), 256, 0, stream>>>(
        x, gamma, beta, Qb, Kb, VTb, out);
}
